// Attention_86260123173325
// MI455X (gfx1250) — matrix operand order repaired and checked by reading, not compiled, not run on hardware
//
#include <hip/hip_runtime.h>

// ---------- WMMA vector types (CDNA5 / gfx1250, wave32) ----------
typedef __attribute__((ext_vector_type(16))) __bf16 v16bf;
typedef __attribute__((ext_vector_type(8)))  float  v8f;
typedef __attribute__((ext_vector_type(4)))  int    v4i;

union BF16x16 { v16bf v; unsigned int u[8]; };
static_assert(sizeof(v16bf) == 32, "v16bf must be 8 VGPRs");

// ---------- CDNA5 async global->LDS copy (probe-confirmed builtin) ----------
#if __has_builtin(__builtin_amdgcn_global_load_async_to_lds_b128)
#define HAS_ASYNC_LDS 1
typedef __attribute__((address_space(1))) v4i GV4;   // global int4*
typedef __attribute__((address_space(3))) v4i LV4;   // LDS int4*
#else
#define HAS_ASYNC_LDS 0
#endif

__device__ __forceinline__ void async_wait0() {
#if HAS_ASYNC_LDS
#if __has_builtin(__builtin_amdgcn_s_wait_asynccnt)
    __builtin_amdgcn_s_wait_asynccnt(0);
#else
    asm volatile("s_wait_asynccnt 0" ::: "memory");
#endif
#endif
}

__device__ __forceinline__ void copy16B_to_lds(const __bf16* src, __bf16* dst) {
#if HAS_ASYNC_LDS
    // generic->AS1 is address-identity; generic->AS3 is the low-32-bit
    // LDS offset (AMDGPU flat->local addrspacecast == trunc to i32).
    GV4* g = (GV4*)(unsigned long long)(uintptr_t)src;
    LV4* l = (LV4*)(unsigned int)(uintptr_t)dst;
    __builtin_amdgcn_global_load_async_to_lds_b128(g, l, 0, 0);
#else
    *reinterpret_cast<float4*>(dst) = *reinterpret_cast<const float4*>(src);
#endif
}

// =================================================================
// Kernel 1: convert q (pre-scaled by 0.5) and k to bf16, row-major.
// =================================================================
__global__ __launch_bounds__(256) void convert_qk(const float* __restrict__ x,
                                                  __bf16* __restrict__ qh,
                                                  __bf16* __restrict__ kh) {
    size_t i = ((size_t)blockIdx.x * 256 + threadIdx.x) * 8;
    size_t n = i >> 12;
    size_t d = i & 4095;
    const float* qsrc = x + (n * 3 + 0) * 4096 + d;
    const float* ksrc = x + (n * 3 + 1) * 4096 + d;
#pragma unroll
    for (int j = 0; j < 8; ++j) {
        qh[i + j] = (__bf16)(0.5f * qsrc[j]);
        kh[i + j] = (__bf16)(ksrc[j]);
    }
}

// =================================================================
// Kernel 2: v (c2) -> vT bf16 via LDS-tiled transpose (64x64 tiles).
// =================================================================
__global__ __launch_bounds__(256) void transpose_v(const float* __restrict__ x,
                                                   __bf16* __restrict__ vT) {
    __shared__ __bf16 tile[64][65];
    int n0 = blockIdx.y * 64;
    int d0 = blockIdx.x * 64;
    int tx = threadIdx.x & 63;
    int ty = threadIdx.x >> 6;
#pragma unroll
    for (int i = 0; i < 16; ++i) {
        int r = ty * 16 + i;
        tile[r][tx] = (__bf16)x[((size_t)(n0 + r) * 3 + 2) * 4096 + d0 + tx];
    }
    __syncthreads();
#pragma unroll
    for (int i = 0; i < 16; ++i) {
        int r = ty * 16 + i;
        vT[(size_t)(d0 + r) * 4096 + n0 + tx] = tile[tx][r];
    }
}

// =================================================================
// GEMM: C(f32) = A(bf16, row-major) * Brows(bf16)^T, M=N=K=4096.
//   WG tile 256x128, BK=32, 8 waves, each wave 64x64 (4x4 WMMA tiles).
//   Double-buffered LDS stages filled by async global->LDS copies;
//   one workgroup barrier per K-iteration.
// =================================================================
#define LDT     40                     // padded stride: rows 16B-aligned,
#define A_ELEMS (256 * LDT)            // bank-stride 20 -> conflict-free
#define B_ELEMS (128 * LDT)
#define STAGE_ELEMS (A_ELEMS + B_ELEMS)

__global__ __launch_bounds__(256) void gemm_bf16_nt(const __bf16* __restrict__ A,
                                                    const __bf16* __restrict__ Brows,
                                                    float* __restrict__ C) {
    __shared__ __bf16 lds[2][STAGE_ELEMS];   // 60 KB total

    const int tid  = threadIdx.x;
    const int bm   = blockIdx.y;
    const int bn   = blockIdx.x;
    const int wave = tid >> 5;
    const int lane = tid & 31;
    const int wm   = wave & 3;
    const int wn   = wave >> 2;
    const int l    = lane & 15;
    const int hi   = lane >> 4;

    v8f acc[4][4];
#pragma unroll
    for (int mt = 0; mt < 4; ++mt)
#pragma unroll
        for (int nt = 0; nt < 4; ++nt) acc[mt][nt] = (v8f)0.0f;

    const int rr = tid >> 2;
    const int cc = (tid & 3) * 8;
    const __bf16* Agp = &A[(size_t)(bm * 256 + rr) * 4096 + cc];
    const __bf16* Bgp = &Brows[(size_t)(bn * 128 + rr) * 4096 + cc];

    auto stage_copy = [&](int s, int k0) {
        __bf16* Ab = &lds[s][0];
        __bf16* Bb = &lds[s][A_ELEMS];
#pragma unroll
        for (int i = 0; i < 4; ++i) {
            int r = i * 64 + rr;
            copy16B_to_lds(Agp + (size_t)i * 64 * 4096 + k0, &Ab[r * LDT + cc]);
        }
#pragma unroll
        for (int i = 0; i < 2; ++i) {
            int r = i * 64 + rr;
            copy16B_to_lds(Bgp + (size_t)i * 64 * 4096 + k0, &Bb[r * LDT + cc]);
        }
    };

    stage_copy(0, 0);

    int stage = 0;
    for (int k0 = 0; k0 < 4096; k0 += 32) {
        async_wait0();                        // current stage landed (this wave)
        __syncthreads();                      // ..and for every wave in the WG
        if (k0 + 32 < 4096)
            stage_copy(stage ^ 1, k0 + 32);   // async prefetch of next panel

        const __bf16* Ab = &lds[stage][0];
        const __bf16* Bb = &lds[stage][A_ELEMS];

        BF16x16 a[4], b[4];
        // A fragment (ISA 7.12.2): K = (j<4?0:16) + hi*8 + (j&3)*2
#pragma unroll
        for (int mt = 0; mt < 4; ++mt) {
            int row = wm * 64 + mt * 16 + l;
#pragma unroll
            for (int j = 0; j < 8; ++j) {
                int K = ((j < 4) ? 0 : 16) + hi * 8 + (j & 3) * 2;
                a[mt].u[j] = *reinterpret_cast<const unsigned int*>(&Ab[row * LDT + K]);
            }
        }
        // B fragment: K = (j<4?0:16) + hi*8 + (j&3)*2
#pragma unroll
        for (int nt = 0; nt < 4; ++nt) {
            int row = wn * 64 + nt * 16 + l;
#pragma unroll
            for (int j = 0; j < 8; ++j) {
                int K = ((j < 4) ? 0 : 16) + hi * 8 + (j & 3) * 2;
                b[nt].u[j] = *reinterpret_cast<const unsigned int*>(&Bb[row * LDT + K]);
            }
        }
#pragma unroll
        for (int mt = 0; mt < 4; ++mt)
#pragma unroll
            for (int nt = 0; nt < 4; ++nt)
                acc[mt][nt] = __builtin_amdgcn_wmma_f32_16x16x32_bf16(
                    false, a[mt].v, false, b[nt].v,
                    (short)0, acc[mt][nt], false, false);

        stage ^= 1;
    }

#pragma unroll
    for (int mt = 0; mt < 4; ++mt)
#pragma unroll
        for (int nt = 0; nt < 4; ++nt) {
            int col = bn * 128 + wn * 64 + nt * 16 + l;
#pragma unroll
            for (int i = 0; i < 8; ++i) {
                int row = bm * 256 + wm * 64 + mt * 16 + hi * 8 + i;
                C[(size_t)row * 4096 + col] = acc[mt][nt][i];
            }
        }
}

// =================================================================
// Kernel 4: row softmax over dp (fp32, 4096 wide), write bf16 attn.
// =================================================================
__global__ __launch_bounds__(256) void softmax_row(const float* __restrict__ dp,
                                                   __bf16* __restrict__ attn) {
    __shared__ float red[256];
    const int row = blockIdx.x;
    const int t   = threadIdx.x;
    const float* p = dp + (size_t)row * 4096;

    float vals[16];
    float m = -3.0e38f;
#pragma unroll
    for (int i = 0; i < 16; ++i) {
        vals[i] = p[t + i * 256];
        m = fmaxf(m, vals[i]);
    }
    red[t] = m; __syncthreads();
    for (int s = 128; s > 0; s >>= 1) {
        if (t < s) red[t] = fmaxf(red[t], red[t + s]);
        __syncthreads();
    }
    m = red[0]; __syncthreads();

    float sum = 0.0f;
#pragma unroll
    for (int i = 0; i < 16; ++i) {
        vals[i] = __expf(vals[i] - m);
        sum += vals[i];
    }
    red[t] = sum; __syncthreads();
    for (int s = 128; s > 0; s >>= 1) {
        if (t < s) red[t] += red[t + s];
        __syncthreads();
    }
    const float inv = 1.0f / red[0];

    __bf16* o = attn + (size_t)row * 4096;
#pragma unroll
    for (int i = 0; i < 16; ++i)
        o[t + i * 256] = (__bf16)(vals[i] * inv);
}

// =================================================================
// Launch.  ws: [0,32MB) qh (reused as attn) | [32,64MB) kh | [64,96MB) vT
// dp (fp32 4096x4096) lives in d_out (exactly 64 MB).
// =================================================================
extern "C" void kernel_launch(void* const* d_in, const int* in_sizes, int n_in,
                              void* d_out, int out_size, void* d_ws, size_t ws_size,
                              hipStream_t stream) {
    (void)in_sizes; (void)n_in; (void)out_size; (void)ws_size;
    const float* x   = (const float*)d_in[0];
    float*       out = (float*)d_out;
    char*        ws  = (char*)d_ws;

    __bf16* qh = (__bf16*)(ws);
    __bf16* kh = (__bf16*)(ws + ((size_t)32 << 20));
    __bf16* vT = (__bf16*)(ws + ((size_t)64 << 20));
    float*  dp = out;

    convert_qk <<<8192,          256, 0, stream>>>(x, qh, kh);
    transpose_v<<<dim3(64, 64),  256, 0, stream>>>(x, vT);
    gemm_bf16_nt<<<dim3(32, 16), 256, 0, stream>>>(qh, kh, dp);  // dp = (0.5 q) k^T
    softmax_row<<<4096,          256, 0, stream>>>(dp, qh);      // attn -> qh
    gemm_bf16_nt<<<dim3(32, 16), 256, 0, stream>>>(qh, vT, out); // out = attn v
}
